// mBloSA_84293028151835
// MI455X (gfx1250) — compile-verified, matrix operand order checked by reading, not run on hardware
//
#include <hip/hip_runtime.h>
#include <hip/hip_bf16.h>

// mBloSA for MI455X (gfx1250, wave32, WMMA).
//
// Key simplification (exact math of the reference): _msa reduces to
//   out[i] = x[i] * sum_j softmax_j(...) = x[i]  (rows fully masked -> 0),
// so h = x with token (i%32==31) zeroed, and o = v with the last block of
// each batch zeroed. m_W1/m_W2/m_b cancel. What remains is 6 GEMMs + small
// softmax/sigmoid epilogues, all driven by v_wmma_f32_16x16x32_bf16 with
// fp32 accumulation.

typedef __bf16 bf16_t;
typedef __attribute__((ext_vector_type(16))) __bf16 v16bf;
typedef __attribute__((ext_vector_type(8)))  __bf16 v8bf;
typedef __attribute__((ext_vector_type(4)))  __bf16 v4bf;
typedef __attribute__((ext_vector_type(8)))  float  v8f;

#define TOK  16384   // B * n  (4 * 4096) token rows
#define DW   256     // feature dim D
#define MBLK 512     // B * m  (4 * 128) block rows
#define KCAT 768     // 3 * D concat width

// ---------------------------------------------------------------------------
// Fragment load: CDNA5 16-bit A/B 16x32 layout (wave32).
//   lane l<16 : row M=l, holds K = {0..7} and {16..23}
//   lane l>=16: row M=l-16, holds K = {8..15} and {24..31}
// rowptr must already point at element (row = lane&15, k0) of a row-major
// matrix; this adds the lane-dependent k subgroup offset. Two b128 loads.
// ---------------------------------------------------------------------------
__device__ __forceinline__ v16bf load_frag_row(const bf16_t* __restrict__ rowptr,
                                               int lane) {
  const bf16_t* p = rowptr + ((lane >> 4) << 3);   // +0 or +8
  v8bf lo = *(const v8bf*)(p);                     // K = koff .. koff+7
  v8bf hi = *(const v8bf*)(p + 16);                // K = koff+16 .. koff+23
  return __builtin_shufflevector(lo, hi, 0, 1, 2, 3, 4, 5, 6, 7,
                                         8, 9, 10, 11, 12, 13, 14, 15);
}

__device__ __forceinline__ v8f wmma_bf16(v16bf a, v16bf b, v8f c) {
  return __builtin_amdgcn_wmma_f32_16x16x32_bf16(
      /*neg_a=*/false, a, /*neg_b=*/false, b,
      /*c_mod=*/(short)0, c, /*reuse_a=*/false, /*reuse_b=*/false);
}

// ---------------------------------------------------------------------------
// fp32 -> bf16 weight conversion
// ---------------------------------------------------------------------------
__global__ void cvt_bf16_kernel(const float* __restrict__ s,
                                bf16_t* __restrict__ d, int n) {
  int i = blockIdx.x * blockDim.x + threadIdx.x;
  if (i < n) d[i] = (bf16_t)s[i];
}

// ---------------------------------------------------------------------------
// x -> x_bf, h_bf = x_bf with token (tok%32 == 31) zeroed (the _msa identity)
// one float4 per thread (64 float4 per 256-wide row)
// ---------------------------------------------------------------------------
__global__ void pack_xh_kernel(const float4* __restrict__ x4,
                               v4bf* __restrict__ xb4,
                               v4bf* __restrict__ hb4) {
  int i = blockIdx.x * blockDim.x + threadIdx.x;   // 0 .. TOK*64-1
  float4 v = x4[i];
  v4bf b;
  b[0] = (bf16_t)v.x; b[1] = (bf16_t)v.y; b[2] = (bf16_t)v.z; b[3] = (bf16_t)v.w;
  xb4[i] = b;
  int tok = i >> 6;
  if ((tok & 31) == 31) { v4bf z = {}; hb4[i] = z; } else { hb4[i] = b; }
}

// ---------------------------------------------------------------------------
// Generic GEMM: out = epi(A[M,K] @ W[N,K]^T + bias).  One 16x16 tile per wave.
//   EPI 0: fp32 store  (s2t scores)
//   EPI 1: relu -> bf16 store (s2t hidden)
// ---------------------------------------------------------------------------
template <int EPI>
__global__ void gemm_bf16_kernel(const bf16_t* __restrict__ A,
                                 const bf16_t* __restrict__ W,
                                 const float* __restrict__ bias,
                                 void* __restrict__ out,
                                 int M, int N, int K) {
  int lane = threadIdx.x & 31;
  int tile = blockIdx.x * (blockDim.x >> 5) + (threadIdx.x >> 5);
  int tilesN = N >> 4;
  int tr = tile / tilesN;
  int tm = tr << 4;
  int tn = (tile - tr * tilesN) << 4;
  if (tm >= M) return;

  const bf16_t* Arow = A + (size_t)(tm + (lane & 15)) * K;
  const bf16_t* Wrow = W + (size_t)(tn + (lane & 15)) * K;

  v8f acc = {};
  for (int k0 = 0; k0 < K; k0 += 32) {
    v16bf a = load_frag_row(Arow + k0, lane);
    v16bf b = load_frag_row(Wrow + k0, lane);
    acc = wmma_bf16(a, b, acc);
  }

  int col  = tn + (lane & 15);
  int row0 = tm + ((lane >> 4) << 3);
  float bz = bias[col];
#pragma unroll
  for (int i = 0; i < 8; ++i) {
    float v = acc[i] + bz;
    size_t idx = (size_t)(row0 + i) * N + col;
    if (EPI == 0) ((float*)out)[idx] = v;
    else          ((bf16_t*)out)[idx] = (bf16_t)fmaxf(v, 0.0f);
  }
}

// ---------------------------------------------------------------------------
// s2t: per 32-token block, per feature d:  softmax over the 32 rows of
// `scores`, then v[blk,d] = sum_i p_i * h[i,d]  (h row 31 is zero).
// One workgroup per block (512), one thread per feature (256). Coalesced.
// ---------------------------------------------------------------------------
__global__ void s2t_softmax_kernel(const float* __restrict__ scores,
                                   const float* __restrict__ x,
                                   float* __restrict__ v_f32,
                                   bf16_t* __restrict__ v_bf,
                                   bf16_t* __restrict__ o_bf) {
  int blk = blockIdx.x;        // 0..511 (batch*128 + block)
  int d   = threadIdx.x;       // 0..255
  const float* S = scores + (size_t)blk * 32 * DW + d;
  const float* X = x      + (size_t)blk * 32 * DW + d;

  float mx = -1e30f;
#pragma unroll
  for (int i = 0; i < 32; ++i) mx = fmaxf(mx, S[i * DW]);
  float sum = 0.0f, acc = 0.0f;
#pragma unroll
  for (int i = 0; i < 32; ++i) {
    float p = __expf(S[i * DW] - mx);
    sum += p;
    float h = (i == 31) ? 0.0f : X[i * DW];   // h = masked x (msa identity)
    acc += p * h;
  }
  float val = acc / sum;
  size_t idx = (size_t)blk * DW + d;
  v_f32[idx] = val;
  v_bf[idx]  = (bf16_t)val;
  // o = v with last block of each batch zeroed (second msa identity)
  o_bf[idx]  = ((blk & 127) == 127) ? (bf16_t)0.0f : (bf16_t)val;
}

// ---------------------------------------------------------------------------
// Gate: G = sigmoid(o@g_W1^T + v@g_W2^T + g_b); e = G*o + (1-G)*v
// Dual accumulators, one 16x16 tile per wave, M=512 N=256 K=256.
// ---------------------------------------------------------------------------
__global__ void gate_kernel(const bf16_t* __restrict__ o_bf,
                            const bf16_t* __restrict__ v_bf,
                            const bf16_t* __restrict__ W1,
                            const bf16_t* __restrict__ W2,
                            const float* __restrict__ gb,
                            const float* __restrict__ v_f32,
                            bf16_t* __restrict__ e_bf) {
  int lane = threadIdx.x & 31;
  int tile = blockIdx.x * (blockDim.x >> 5) + (threadIdx.x >> 5);
  int tm = (tile >> 4) << 4;           // tilesN = 256/16 = 16
  int tn = (tile & 15) << 4;

  const bf16_t* Orow = o_bf + (size_t)(tm + (lane & 15)) * DW;
  const bf16_t* Vrow = v_bf + (size_t)(tm + (lane & 15)) * DW;
  const bf16_t* W1r  = W1   + (size_t)(tn + (lane & 15)) * DW;
  const bf16_t* W2r  = W2   + (size_t)(tn + (lane & 15)) * DW;

  v8f acc1 = {}, acc2 = {};
#pragma unroll
  for (int k0 = 0; k0 < DW; k0 += 32) {
    v16bf ao = load_frag_row(Orow + k0, lane);
    v16bf av = load_frag_row(Vrow + k0, lane);
    v16bf b1 = load_frag_row(W1r + k0, lane);
    v16bf b2 = load_frag_row(W2r + k0, lane);
    acc1 = wmma_bf16(ao, b1, acc1);
    acc2 = wmma_bf16(av, b2, acc2);
  }

  int col  = tn + (lane & 15);
  int row0 = tm + ((lane >> 4) << 3);
  float bz = gb[col];
#pragma unroll
  for (int i = 0; i < 8; ++i) {
    int row = row0 + i;
    float vv = v_f32[(size_t)row * DW + col];
    float oo = ((row & 127) == 127) ? 0.0f : vv;
    float g  = 1.0f / (1.0f + __expf(-(acc1[i] + acc2[i] + bz)));
    e_bf[(size_t)row * DW + col] = (bf16_t)(g * oo + (1.0f - g) * vv);
  }
}

// ---------------------------------------------------------------------------
// Fusion: cat = [x, h, E] assembled on the fly (E broadcast via row>>5),
// fusion = relu(cat@f_W1^T + b1), G2 = sigmoid(cat@f_W2^T + b2),
// out = G2*fusion + (1-G2)*x.   M=16384 N=256 K=768, dual accumulators.
// ---------------------------------------------------------------------------
__global__ void fusion_kernel(const bf16_t* __restrict__ x_bf,
                              const bf16_t* __restrict__ h_bf,
                              const bf16_t* __restrict__ e_bf,
                              const bf16_t* __restrict__ W1,
                              const bf16_t* __restrict__ W2,
                              const float* __restrict__ b1,
                              const float* __restrict__ b2,
                              const float* __restrict__ x_f32,
                              float* __restrict__ out) {
  int lane = threadIdx.x & 31;
  int tile = blockIdx.x * (blockDim.x >> 5) + (threadIdx.x >> 5);
  int tm = (tile >> 4) << 4;           // tilesN = 16
  int tn = (tile & 15) << 4;

  int arow = tm + (lane & 15);
  const bf16_t* a0 = x_bf + (size_t)arow * DW;          // k in [0,256)
  const bf16_t* a1 = h_bf + (size_t)arow * DW;          // k in [256,512)
  const bf16_t* a2 = e_bf + (size_t)(arow >> 5) * DW;   // k in [512,768)
  const bf16_t* W1r = W1 + (size_t)(tn + (lane & 15)) * KCAT;
  const bf16_t* W2r = W2 + (size_t)(tn + (lane & 15)) * KCAT;

  v8f acc1 = {}, acc2 = {};
#pragma unroll
  for (int k0 = 0; k0 < KCAT; k0 += 32) {
    const bf16_t* ar = (k0 < DW)     ? (a0 + k0)
                     : (k0 < 2 * DW) ? (a1 + (k0 - DW))
                                     : (a2 + (k0 - 2 * DW));
    v16bf a   = load_frag_row(ar, lane);
    v16bf bb1 = load_frag_row(W1r + k0, lane);
    v16bf bb2 = load_frag_row(W2r + k0, lane);
    acc1 = wmma_bf16(a, bb1, acc1);
    acc2 = wmma_bf16(a, bb2, acc2);
  }

  int col  = tn + (lane & 15);
  int row0 = tm + ((lane >> 4) << 3);
  float bz1 = b1[col], bz2 = b2[col];
#pragma unroll
  for (int i = 0; i < 8; ++i) {
    int row = row0 + i;
    float fus = fmaxf(acc1[i] + bz1, 0.0f);
    float g2  = 1.0f / (1.0f + __expf(-(acc2[i] + bz2)));
    float xv  = x_f32[(size_t)row * DW + col];
    out[(size_t)row * DW + col] = g2 * fus + (1.0f - g2) * xv;
  }
}

// ---------------------------------------------------------------------------
// Host orchestration
// ---------------------------------------------------------------------------
extern "C" void kernel_launch(void* const* d_in, const int* in_sizes, int n_in,
                              void* d_out, int out_size, void* d_ws, size_t ws_size,
                              hipStream_t stream) {
  const float* x        = (const float*)d_in[0];
  const float* s2t_W1_w = (const float*)d_in[1];
  const float* s2t_W1_b = (const float*)d_in[2];
  const float* s2t_W_w  = (const float*)d_in[3];
  const float* s2t_W_b  = (const float*)d_in[4];
  // d_in[5..7]: m_W1, m_W2, m_b — mathematically cancel in _msa (see header)
  const float* g_W1     = (const float*)d_in[8];
  const float* g_W2     = (const float*)d_in[9];
  const float* g_b      = (const float*)d_in[10];
  const float* f_W1_w   = (const float*)d_in[11];
  const float* f_W1_b   = (const float*)d_in[12];
  const float* f_W2_w   = (const float*)d_in[13];
  const float* f_W2_b   = (const float*)d_in[14];

  // workspace carve (256B aligned slices)
  char* p = (char*)d_ws;
  auto take = [&](size_t bytes) -> char* {
    char* r = p;
    p += (bytes + 255) & ~(size_t)255;
    return r;
  };
  bf16_t* x_bf    = (bf16_t*)take((size_t)TOK * DW * 2);
  bf16_t* h_bf    = (bf16_t*)take((size_t)TOK * DW * 2);
  bf16_t* t1_bf   = (bf16_t*)take((size_t)TOK * DW * 2);
  float*  scores  = (float*) take((size_t)TOK * DW * 4);
  float*  v_f32   = (float*) take((size_t)MBLK * DW * 4);
  bf16_t* v_bf    = (bf16_t*)take((size_t)MBLK * DW * 2);
  bf16_t* o_bf    = (bf16_t*)take((size_t)MBLK * DW * 2);
  bf16_t* e_bf    = (bf16_t*)take((size_t)MBLK * DW * 2);
  bf16_t* w_s2t1  = (bf16_t*)take((size_t)DW * DW * 2);
  bf16_t* w_s2t   = (bf16_t*)take((size_t)DW * DW * 2);
  bf16_t* w_g1    = (bf16_t*)take((size_t)DW * DW * 2);
  bf16_t* w_g2    = (bf16_t*)take((size_t)DW * DW * 2);
  bf16_t* w_f1    = (bf16_t*)take((size_t)DW * KCAT * 2);
  bf16_t* w_f2    = (bf16_t*)take((size_t)DW * KCAT * 2);

  // 1) weights fp32 -> bf16
  cvt_bf16_kernel<<<256, 256, 0, stream>>>(s2t_W1_w, w_s2t1, DW * DW);
  cvt_bf16_kernel<<<256, 256, 0, stream>>>(s2t_W_w,  w_s2t,  DW * DW);
  cvt_bf16_kernel<<<256, 256, 0, stream>>>(g_W1,     w_g1,   DW * DW);
  cvt_bf16_kernel<<<256, 256, 0, stream>>>(g_W2,     w_g2,   DW * DW);
  cvt_bf16_kernel<<<768, 256, 0, stream>>>(f_W1_w,   w_f1,   DW * KCAT);
  cvt_bf16_kernel<<<768, 256, 0, stream>>>(f_W2_w,   w_f2,   DW * KCAT);

  // 2) x -> bf16, h = masked x (msa identity)
  pack_xh_kernel<<<(TOK * (DW / 4)) / 256, 256, 0, stream>>>(
      (const float4*)x, (v4bf*)x_bf, (v4bf*)h_bf);

  // 3) s2t hidden: t1 = relu(h @ s2t_W1^T + b)   [16384x256x256]
  gemm_bf16_kernel<1><<<2048, 256, 0, stream>>>(h_bf, w_s2t1, s2t_W1_b,
                                                (void*)t1_bf, TOK, DW, DW);
  // 4) s2t scores: scores = t1 @ s2t_W^T + b     [16384x256x256]
  gemm_bf16_kernel<0><<<2048, 256, 0, stream>>>(t1_bf, w_s2t, s2t_W_b,
                                                (void*)scores, TOK, DW, DW);
  // 5) per-block softmax + weighted sum -> v, o
  s2t_softmax_kernel<<<MBLK, DW, 0, stream>>>(scores, x, v_f32, v_bf, o_bf);

  // 6) gate -> e                                  [512x256x256, dual]
  gate_kernel<<<64, 256, 0, stream>>>(o_bf, v_bf, w_g1, w_g2, g_b, v_f32, e_bf);

  // 7) fusion -> out                              [16384x256x768, dual]
  fusion_kernel<<<2048, 256, 0, stream>>>(x_bf, h_bf, e_bf, w_f1, w_f2,
                                          f_W1_b, f_W2_b, x, (float*)d_out);
}
